// Fuser_Model_38878043963454
// MI455X (gfx1250) — compile-verified, matrix operand order checked by reading, not run on hardware
//
#include <hip/hip_runtime.h>
#include <math.h>

// ---------------------------------------------------------------------------
// CDNA5 (gfx1250) VMamba/StyleGAN "fuser".
// All dense contractions go through one bf16 WMMA GEMM whose operands are
// PRE-PACKED into the ISA 7.12.2 fragment layout:
//   P[(kb*Rp + r)*16 + p] = bf16pair( M[r][kb*32+2p], M[r][kb*32+2p+1] )
// so each lane's 8 fragment VGPRs are two contiguous b128 loads. Packers
// zero-pad (M->16, N->64, K->32): the GEMM K-loop has no guards at all, and
// is double-buffered so global loads overlap WMMA issue.
// ---------------------------------------------------------------------------

typedef __attribute__((ext_vector_type(16))) __bf16   v16bf;
typedef __attribute__((ext_vector_type(8)))  float    v8f;
typedef __attribute__((ext_vector_type(4)))  unsigned u32x4;

__device__ __forceinline__ float siluf(float x)     { return x / (1.0f + __expf(-x)); }
__device__ __forceinline__ float softplusf(float x) { return (x > 20.0f) ? x : log1pf(__expf(x)); }
__device__ __forceinline__ unsigned short f2bf(float f) {
  unsigned u = __float_as_uint(f);
  u += 0x7FFFu + ((u >> 16) & 1u);          // round-to-nearest-even
  return (unsigned short)(u >> 16);
}
__device__ __forceinline__ int imin(int a, int b) { return a < b ? a : b; }
__device__ __forceinline__ int imax(int a, int b) { return a > b ? a : b; }

union FragU { u32x4 q[2]; unsigned u[8]; v16bf v; };

__device__ __forceinline__ v8f wmma_bf16(const FragU& a, const FragU& b, v8f c) {
  return __builtin_amdgcn_wmma_f32_16x16x32_bf16(false, a.v, false, b.v, (short)0, c, false, false);
}

// ---------------------------------------------------------------------------
// WMMA GEMM on packed operands. D[M,N] = act(A*B + bias[M] (+D)).
// Wave computes a 16x64 strip: 2 b128 A loads + 8 b128 B loads + 4 WMMA / K-step,
// two-stage double buffered (prefetch K-step kb+1 while WMMAing kb).
// ---------------------------------------------------------------------------
__global__ void k_gemm(const unsigned* __restrict__ Ap, const unsigned* __restrict__ Bp,
                       const float* __restrict__ bias, float* __restrict__ D,
                       int M, int N, int ldd, int Mp, int Np, int nkb,
                       int accum, int act) {
  int lane = threadIdx.x;
  int row  = lane & 15, hi = lane >> 4;
  int tm = blockIdx.y * 16;
  int tn = blockIdx.x * 64;

  unsigned ao    = (unsigned)(tm + row) * 16u + (unsigned)hi * 4u;
  unsigned bo    = (unsigned)(tn + row) * 16u + (unsigned)hi * 4u;
  unsigned astep = (unsigned)Mp * 16u;
  unsigned bstep = (unsigned)Np * 16u;

  const v8f vzero = {0.f, 0.f, 0.f, 0.f, 0.f, 0.f, 0.f, 0.f};
  v8f acc[4];
#pragma unroll
  for (int j = 0; j < 4; ++j) acc[j] = vzero;

  FragU fa0, fa1;
  FragU fb0[4], fb1[4];

#define LDA(f, o)  do { (f).q[0] = *(const u32x4*)(Ap + (o)); \
                        (f).q[1] = *(const u32x4*)(Ap + (o) + 8); } while (0)
#define LDB(f, o)  do { _Pragma("unroll") \
                        for (int j = 0; j < 4; ++j) { \
                          (f)[j].q[0] = *(const u32x4*)(Bp + (o) + j * 256); \
                          (f)[j].q[1] = *(const u32x4*)(Bp + (o) + j * 256 + 8); } } while (0)

  LDA(fa0, ao); LDB(fb0, bo); ao += astep; bo += bstep;

  int kb = 0;
  for (; kb + 1 < nkb; kb += 2) {
    LDA(fa1, ao); LDB(fb1, bo); ao += astep; bo += bstep;
#pragma unroll
    for (int j = 0; j < 4; ++j) acc[j] = wmma_bf16(fa0, fb0[j], acc[j]);
    if (kb + 2 < nkb) { LDA(fa0, ao); LDB(fb0, bo); ao += astep; bo += bstep; }
#pragma unroll
    for (int j = 0; j < 4; ++j) acc[j] = wmma_bf16(fa1, fb1[j], acc[j]);
  }
  if (kb < nkb) {
#pragma unroll
    for (int j = 0; j < 4; ++j) acc[j] = wmma_bf16(fa0, fb0[j], acc[j]);
  }
#undef LDA
#undef LDB

#pragma unroll
  for (int j = 0; j < 4; ++j) {
    int bn = tn + 16 * j + row;
    if (bn < N) {
#pragma unroll
      for (int rr = 0; rr < 8; ++rr) {
        int m = tm + rr + 8 * hi;
        if (m < M) {
          float val = acc[j][rr];
          if (bias)  val += bias[m];
          if (accum) val += D[(size_t)m * ldd + bn];
          if (act == 1) val = siluf(val);
          D[(size_t)m * ldd + bn] = val;
        }
      }
    }
  }
}

// ---------------------------------------------------------------------------
// Fragment packers (fp32 -> packed bf16 pairs, zero-padded)
// ---------------------------------------------------------------------------
// A-side: rows of A ([M,K] row-major, leading dim ld)
__global__ void k_pack_a(const float* __restrict__ src, unsigned* __restrict__ dst,
                         int M, int K, int ld, int Mp, int Kp) {
  long i = (long)blockIdx.x * blockDim.x + threadIdx.x;
  long total = (long)Mp * (Kp / 2);
  if (i >= total) return;
  int p  = (int)(i & 15);
  int m  = (int)((i >> 4) % Mp);
  int kb = (int)(i / ((long)Mp * 16));
  int k  = kb * 32 + 2 * p;
  unsigned lo = 0, hi = 0;
  if (m < M) {
    if (k < K)     lo = f2bf(src[(size_t)m * ld + k]);
    if (k + 1 < K) hi = f2bf(src[(size_t)m * ld + k + 1]);
  }
  dst[i] = lo | (hi << 16);
}

// B-side: columns of B ([K,N] row-major, leading dim ld)
__global__ void k_pack_b(const float* __restrict__ src, unsigned* __restrict__ dst,
                         int N, int K, int ld, int Np, int Kp) {
  long i = (long)blockIdx.x * blockDim.x + threadIdx.x;
  long total = (long)Np * (Kp / 2);
  if (i >= total) return;
  int p  = (int)(i & 15);
  int n  = (int)((i >> 4) % Np);
  int kb = (int)(i / ((long)Np * 16));
  int k  = kb * 32 + 2 * p;
  unsigned lo = 0, hi = 0;
  if (n < N) {
    if (k < K)     lo = f2bf(src[(size_t)k * ld + n]);
    if (k + 1 < K) hi = f2bf(src[(size_t)(k + 1) * ld + n]);
  }
  dst[i] = lo | (hi << 16);
}

// im2col (3x3, pad 1) emitting packed B directly; K = C*9, N = H*W
__global__ void k_im2col_pack(const float* __restrict__ x, unsigned* __restrict__ dst,
                              int C, int H, int W, int Np, int Kp) {
  long i = (long)blockIdx.x * blockDim.x + threadIdx.x;
  long total = (long)Np * (Kp / 2);
  if (i >= total) return;
  int p  = (int)(i & 15);
  int n  = (int)((i >> 4) % Np);
  int kb = (int)(i / ((long)Np * 16));
  int K  = C * 9;
  long HW = (long)H * W;
  int ho = n / W, wo = n % W;
  unsigned half[2] = {0u, 0u};
#pragma unroll
  for (int e = 0; e < 2; ++e) {
    int k = kb * 32 + 2 * p + e;
    if (n < (int)HW && k < K) {
      int ci = k / 9, rem = k % 9;
      int kh = rem / 3, kw = rem % 3;
      int ih = ho + kh - 1, iw = wo + kw - 1;
      float v = (ih >= 0 && ih < H && iw >= 0 && iw < W) ? x[(size_t)ci * HW + ih * W + iw] : 0.f;
      half[e] = f2bf(v);
    }
  }
  dst[i] = half[0] | (half[1] << 16);
}

// ---------------------------------------------------------------------------
// Elementwise / small kernels
// ---------------------------------------------------------------------------
// dst[c*R + r] = src[r*S + off + c]   (transpose of an R x C view)
__global__ void k_transpose(const float* __restrict__ src, float* __restrict__ dst,
                            int R, int C, int S, int off) {
  long i = (long)blockIdx.x * blockDim.x + threadIdx.x;
  if (i >= (long)R * C) return;
  int r = (int)(i / C), c = (int)(i % C);
  dst[(size_t)c * R + r] = src[(size_t)r * S + off + c];
}

__global__ void k_mul_silu_strided(float* __restrict__ y, int ldy,
                                   const float* __restrict__ z, int ldz,
                                   int rows, int cols) {
  long i = (long)blockIdx.x * blockDim.x + threadIdx.x;
  if (i >= (long)rows * cols) return;
  int r = (int)(i / cols), c = (int)(i % cols);
  y[(size_t)r * ldy + c] *= siluf(z[(size_t)r * ldz + c]);
}

__global__ void k_softplus_gather(const float* __restrict__ src, int lds_,
                                  const float* __restrict__ bias,
                                  float* __restrict__ dst, int rows, int cols) {
  long i = (long)blockIdx.x * blockDim.x + threadIdx.x;
  if (i >= (long)rows * cols) return;
  int r = (int)(i / cols), c = (int)(i % cols);
  dst[i] = softplusf(src[(size_t)r * lds_ + c] + bias[c]);
}

__global__ void k_softplus_rowbias(float* __restrict__ x, const float* __restrict__ bias,
                                   int rows, int cols) {
  long i = (long)blockIdx.x * blockDim.x + threadIdx.x;
  if (i >= (long)rows * cols) return;
  int r = (int)(i / cols);
  x[i] = softplusf(x[i] + bias[r]);
}

// LayerNorm over last dim; one block per row; safe in-place.
__global__ void k_layernorm(const float* __restrict__ x, const float* __restrict__ w,
                            const float* __restrict__ b, float* __restrict__ y, int C) {
  int row = blockIdx.x;
  const float* xr = x + (size_t)row * C;
  float* yr = y + (size_t)row * C;
  __shared__ float red[256];
  float s = 0.f, s2 = 0.f;
  for (int c = threadIdx.x; c < C; c += blockDim.x) { float v = xr[c]; s += v; s2 += v * v; }
  red[threadIdx.x] = s; __syncthreads();
  for (int o = 128; o > 0; o >>= 1) { if (threadIdx.x < o) red[threadIdx.x] += red[threadIdx.x + o]; __syncthreads(); }
  float mean = red[0] / C; __syncthreads();
  red[threadIdx.x] = s2; __syncthreads();
  for (int o = 128; o > 0; o >>= 1) { if (threadIdx.x < o) red[threadIdx.x] += red[threadIdx.x + o]; __syncthreads(); }
  float var = red[0] / C - mean * mean;
  float inv = rsqrtf(var + 1e-5f);
  for (int c = threadIdx.x; c < C; c += blockDim.x)
    yr[c] = (xr[c] - mean) * inv * w[c] + b[c];
}

// ---------------------------------------------------------------------------
// Convolutions (non-GEMM ones)
// ---------------------------------------------------------------------------
__global__ void k_conv1d(const float* __restrict__ x, const float* __restrict__ w,
                         const float* __restrict__ bias, float* __restrict__ y,
                         int Cin, int Cout, int L, int ks, int pad) {
  long i = (long)blockIdx.x * blockDim.x + threadIdx.x;
  if (i >= (long)Cout * L) return;
  int co = (int)(i / L), l = (int)(i % L);
  float acc = bias[co];
  for (int ci = 0; ci < Cin; ++ci)
    for (int j = 0; j < ks; ++j) {
      int ls = l - pad + j;
      if (ls >= 0 && ls < L)
        acc += x[(size_t)ci * L + ls] * w[((size_t)co * Cin + ci) * ks + j];
    }
  y[i] = acc;
}

__global__ void k_dwconv3x3(const float* __restrict__ x, const float* __restrict__ w,
                            const float* __restrict__ bias, float* __restrict__ y,
                            int C, int H, int W, int act) {
  long i = (long)blockIdx.x * blockDim.x + threadIdx.x;
  if (i >= (long)C * H * W) return;
  int wo = (int)(i % W), ho = (int)((i / W) % H), c = (int)(i / ((long)H * W));
  const float* xc = x + (size_t)c * H * W;
  const float* wc = w + (size_t)c * 9;
  float acc = bias[c];
  for (int kh = 0; kh < 3; ++kh)
    for (int kw = 0; kw < 3; ++kw) {
      int ih = ho + kh - 1, iw = wo + kw - 1;
      if (ih >= 0 && ih < H && iw >= 0 && iw < W)
        acc += xc[ih * W + iw] * wc[kh * 3 + kw];
    }
  y[i] = act ? siluf(acc) : acc;
}

// grouped conv for approximator a1: groups=G, 2 in-ch per group, 1 out-ch, 3x3
__global__ void k_groupconv2_3x3(const float* __restrict__ x, const float* __restrict__ w,
                                 const float* __restrict__ bias, float* __restrict__ y,
                                 int G, int H, int W) {
  long i = (long)blockIdx.x * blockDim.x + threadIdx.x;
  if (i >= (long)G * H * W) return;
  int wo = (int)(i % W), ho = (int)((i / W) % H), g = (int)(i / ((long)H * W));
  float acc = bias[g];
  for (int ic = 0; ic < 2; ++ic) {
    const float* xc = x + (size_t)(2 * g + ic) * H * W;
    const float* wc = w + ((size_t)g * 2 + ic) * 9;
    for (int kh = 0; kh < 3; ++kh)
      for (int kw = 0; kw < 3; ++kw) {
        int ih = ho + kh - 1, iw = wo + kw - 1;
        if (ih >= 0 && ih < H && iw >= 0 && iw < W)
          acc += xc[ih * W + iw] * wc[kh * 3 + kw];
      }
  }
  y[i] = acc;
}

// half-pixel bilinear resize (jax.image.resize 'bilinear')
__global__ void k_resize(const float* __restrict__ x, float* __restrict__ y,
                         int C, int Hi, int Wi, int Ho, int Wo) {
  long i = (long)blockIdx.x * blockDim.x + threadIdx.x;
  if (i >= (long)C * Ho * Wo) return;
  int wo = (int)(i % Wo), ho = (int)((i / Wo) % Ho), c = (int)(i / ((long)Ho * Wo));
  float sy = (ho + 0.5f) * (float)Hi / (float)Ho - 0.5f;
  float sx = (wo + 0.5f) * (float)Wi / (float)Wo - 0.5f;
  int y0 = (int)floorf(sy), x0 = (int)floorf(sx);
  float fy = sy - y0, fx = sx - x0;
  int y0c = imin(imax(y0, 0), Hi - 1), y1c = imin(imax(y0 + 1, 0), Hi - 1);
  int x0c = imin(imax(x0, 0), Wi - 1), x1c = imin(imax(x0 + 1, 0), Wi - 1);
  const float* xc = x + (size_t)c * Hi * Wi;
  float v00 = xc[y0c * Wi + x0c], v01 = xc[y0c * Wi + x1c];
  float v10 = xc[y1c * Wi + x0c], v11 = xc[y1c * Wi + x1c];
  y[i] = (1.f - fy) * ((1.f - fx) * v00 + fx * v01) + fy * ((1.f - fx) * v10 + fx * v11);
}

// ---------------------------------------------------------------------------
// SS2D helpers
// ---------------------------------------------------------------------------
__global__ void k_build_xs(const float* __restrict__ xx, float* __restrict__ xs,
                           int d, int H, int W) {
  long i = (long)blockIdx.x * blockDim.x + threadIdx.x;
  long L = (long)H * W;
  if (i >= (long)d * L) return;
  int c = (int)(i / L), l = (int)(i % L);
  size_t cl = (size_t)c * L;
  float v0 = xx[cl + l];
  int wq = l / H, hq = l % H;
  float v1 = xx[cl + hq * W + wq];
  int lf = (int)L - 1 - l;
  float v2 = xx[cl + lf];
  int wq2 = lf / H, hq2 = lf % H;
  float v3 = xx[cl + hq2 * W + wq2];
  size_t stride = (size_t)d * L;
  xs[0 * stride + cl + l] = v0;
  xs[1 * stride + cl + l] = v1;
  xs[2 * stride + cl + l] = v2;
  xs[3 * stride + cl + l] = v3;
}

__global__ void k_combine_ys(const float* __restrict__ ys, float* __restrict__ yt,
                             int d, int H, int W) {
  long i = (long)blockIdx.x * blockDim.x + threadIdx.x;
  long L = (long)H * W;
  if (i >= (long)d * L) return;
  int c = (int)(i / L), t = (int)(i % L);
  int h = t / W, w = t % W;
  int lwh = w * H + h;
  size_t stride = (size_t)d * L, cl = (size_t)c * L;
  float v = ys[0 * stride + cl + t] + ys[2 * stride + cl + (L - 1 - t)] +
            ys[1 * stride + cl + lwh] + ys[3 * stride + cl + (L - 1 - lwh)];
  yt[(size_t)t * d + c] = v;
}

// one lane per (direction k, channel di); 16 states in registers
__global__ void k_scan_vss(const float* __restrict__ xs, const float* __restrict__ dt,
                           const float* __restrict__ Alog, const float* __restrict__ xdbl,
                           int cdim, int r, const float* __restrict__ Dp,
                           float* __restrict__ ys, int d, int L) {
  int t = blockIdx.x * blockDim.x + threadIdx.x;
  if (t >= 4 * d) return;
  int k = t / d;
  float A[16], h[16];
#pragma unroll
  for (int n = 0; n < 16; ++n) { A[n] = -__expf(Alog[(size_t)t * 16 + n]); h[n] = 0.f; }
  const float* xp  = xs + (size_t)t * L;
  const float* dtp = dt + (size_t)t * L;
  const float* Bp  = xdbl + ((size_t)k * cdim + r) * L;
  const float* Cp  = Bp + 16 * (size_t)L;
  float Dv = Dp[t];
  for (int l = 0; l < L; ++l) {
    float dtv = dtp[l], xv = xp[l], dx = dtv * xv, y = 0.f;
#pragma unroll
    for (int n = 0; n < 16; ++n) {
      h[n] = h[n] * __expf(dtv * A[n]) + dx * Bp[(size_t)n * L + l];
      y += h[n] * Cp[(size_t)n * L + l];
    }
    ys[(size_t)t * L + l] = y + Dv * xv;
  }
}

// ---------------------------------------------------------------------------
// Mamba2 (SSD) pieces
// ---------------------------------------------------------------------------
__global__ void k_mamba_conv(const float* __restrict__ zx, const float* __restrict__ w,
                             const float* __restrict__ bias, float* __restrict__ out, int L) {
  long i = (long)blockIdx.x * blockDim.x + threadIdx.x;
  if (i >= (long)L * 1152) return;
  int l = (int)(i / 1152), c = (int)(i % 1152);
  float acc = bias[c];
  for (int j = 0; j < 4; ++j) {
    int ls = l - 3 + j;
    if (ls >= 0) acc += zx[(size_t)ls * 2192 + 1024 + c] * w[c * 4 + j];
  }
  out[i] = siluf(acc);
}

// one lane per (head, p); 64 states in registers; L=32 steps
__global__ void k_scan_mamba(const float* __restrict__ xBC, const float* __restrict__ dt,
                             const float* __restrict__ Alog, const float* __restrict__ Dp,
                             float* __restrict__ y, int L) {
  int t = blockIdx.x * blockDim.x + threadIdx.x;
  if (t >= 1024) return;
  int hh = t >> 6, pp = t & 63;
  float A = -__expf(Alog[hh]);
  float Dv = Dp[hh];
  float h[64];
#pragma unroll
  for (int n = 0; n < 64; ++n) h[n] = 0.f;
  for (int l = 0; l < L; ++l) {
    float dtv = dt[l * 16 + hh];
    float xv  = xBC[(size_t)l * 1152 + hh * 64 + pp];
    float dA  = __expf(dtv * A);
    float dx  = dtv * xv;
    float yy  = 0.f;
    const float* Bl = xBC + (size_t)l * 1152 + 1024;
    const float* Cl = Bl + 64;
#pragma unroll
    for (int n = 0; n < 64; ++n) { h[n] = h[n] * dA + dx * Bl[n]; yy += h[n] * Cl[n]; }
    y[(size_t)l * 1024 + hh * 64 + pp] = yy + Dv * xv;
  }
}

__global__ void k_mamba_norm(const float* __restrict__ y, const float* __restrict__ zx,
                             int ldz, const float* __restrict__ w, float* __restrict__ out, int C) {
  int row = blockIdx.x;
  const float* yr = y + (size_t)row * C;
  const float* zr = zx + (size_t)row * ldz;
  float* orow = out + (size_t)row * C;
  __shared__ float red[256];
  float s2 = 0.f;
  for (int c = threadIdx.x; c < C; c += blockDim.x) {
    float t = yr[c] * siluf(zr[c]); s2 += t * t;
  }
  red[threadIdx.x] = s2; __syncthreads();
  for (int o = 128; o > 0; o >>= 1) { if (threadIdx.x < o) red[threadIdx.x] += red[threadIdx.x + o]; __syncthreads(); }
  float inv = rsqrtf(red[0] / C + 1e-5f);
  for (int c = threadIdx.x; c < C; c += blockDim.x) {
    float t = yr[c] * siluf(zr[c]);
    orow[c] = t * inv * w[c];
  }
}

// ---------------------------------------------------------------------------
// Modulated conv helpers & misc
// ---------------------------------------------------------------------------
__global__ void k_style(const float* __restrict__ ld, float* __restrict__ style) {
  int j = blockIdx.x * blockDim.x + threadIdx.x;
  if (j >= 512) return;
  float s = 0.f;
  for (int i = 0; i < 18; ++i) s += ld[i * 512 + j];
  style[j] = s / 18.f;
}

__global__ void k_modscale(const float* __restrict__ style, const float* __restrict__ mw,
                           const float* __restrict__ mb, float* __restrict__ s) {
  int ci = blockIdx.x * blockDim.x + threadIdx.x;
  if (ci >= 512) return;
  float acc = mb[ci];
  for (int j = 0; j < 512; ++j) acc += style[j] * mw[(size_t)ci * 512 + j];
  s[ci] = acc;
}

// demodulated weight -> packed A' (Mp = Cout rounded to 16; here 512)
__global__ void k_modweight(const float* __restrict__ w, const float* __restrict__ s,
                            unsigned* __restrict__ Ap, int Cin, int Mp) {
  int co = blockIdx.x;
  int Kx = Cin * 9;
  int Kp = (Kx + 31) & ~31;
  float scale = rsqrtf((float)Kx);
  __shared__ float red[256];
  __shared__ float demod_s;
  float acc = 0.f;
  for (int i = threadIdx.x; i < Kx; i += blockDim.x) {
    int ci = i / 9;
    float v = scale * w[(size_t)co * Kx + i] * s[ci];
    acc += v * v;
  }
  red[threadIdx.x] = acc; __syncthreads();
  for (int o = 128; o > 0; o >>= 1) { if (threadIdx.x < o) red[threadIdx.x] += red[threadIdx.x + o]; __syncthreads(); }
  if (threadIdx.x == 0) demod_s = rsqrtf(red[0] + 1e-8f);
  __syncthreads();
  float demod = demod_s;
  for (int u2 = threadIdx.x; u2 < Kp / 2; u2 += blockDim.x) {
    int k = 2 * u2;
    unsigned lo = 0, hi = 0;
    if (k < Kx)     lo = f2bf(scale * w[(size_t)co * Kx + k]     * s[k / 9]       * demod);
    if (k + 1 < Kx) hi = f2bf(scale * w[(size_t)co * Kx + k + 1] * s[(k + 1) / 9] * demod);
    int kb = k >> 5, p = (k & 31) >> 1;
    Ap[((size_t)kb * Mp + co) * 16 + p] = lo | (hi << 16);
  }
}

__global__ void k_interleave(const float* __restrict__ f, const float* __restrict__ le,
                             float* __restrict__ out, int C, long HW) {
  long i = (long)blockIdx.x * blockDim.x + threadIdx.x;
  if (i >= (long)C * HW) return;
  int c = (int)(i / HW);
  long p = i % HW;
  out[(size_t)(2 * c) * HW + p]     = f[i];
  out[(size_t)(2 * c + 1) * HW + p] = le[i];
}

// ---------------------------------------------------------------------------
// Host orchestration
// ---------------------------------------------------------------------------
static inline dim3 g1(long n) { return dim3((unsigned)((n + 255) / 256)); }
static inline int rup(int x, int m) { return (x + m - 1) / m * m; }

static inline void gemm(hipStream_t st, const unsigned* Ap, const unsigned* Bp,
                        const float* bias, float* D, int M, int N, int K,
                        int ldd, int accum, int act) {
  int Mp = rup(M, 16), Np = rup(N, 64), nkb = rup(K, 32) / 32;
  dim3 g(Np / 64, (M + 15) / 16);
  k_gemm<<<g, 32, 0, st>>>(Ap, Bp, bias, D, M, N, ldd, Mp, Np, nkb, accum, act);
}
static inline void packA(hipStream_t st, const float* src, unsigned* dst, int M, int K, int ld) {
  int Mp = rup(M, 16), Kp = rup(K, 32);
  k_pack_a<<<g1((long)Mp * Kp / 2), 256, 0, st>>>(src, dst, M, K, ld, Mp, Kp);
}
static inline void packB(hipStream_t st, const float* src, unsigned* dst, int N, int K, int ld) {
  int Np = rup(N, 64), Kp = rup(K, 32);
  k_pack_b<<<g1((long)Np * Kp / 2), 256, 0, st>>>(src, dst, N, K, ld, Np, Kp);
}
static inline void im2colP(hipStream_t st, const float* x, unsigned* dst, int C, int H, int W) {
  int Np = rup(H * W, 64), Kp = rup(C * 9, 32);
  k_im2col_pack<<<g1((long)Np * Kp / 2), 256, 0, st>>>(x, dst, C, H, W, Np, Kp);
}

struct VssBufs {
  float *ln, *xz, *xxT, *xxC, *xs, *xdbl, *dt, *ys, *yt;
  unsigned *aLn, *bIp, *bXs, *aXp, *bDtr, *aDtw, *aYt, *bOp;
};

// VSS block: tok[L,C] += SS2D(LN(tok))
static void run_vss(hipStream_t st, void* const* d_in, int pb, float* tok,
                    int L, int C, int H, int W, const VssBufs& b) {
  const float* LNW = (const float*)d_in[pb + 0];
  const float* LNB = (const float*)d_in[pb + 1];
  const float* IP  = (const float*)d_in[pb + 2];
  const float* CW  = (const float*)d_in[pb + 3];
  const float* CB  = (const float*)d_in[pb + 4];
  const float* XP  = (const float*)d_in[pb + 5];
  const float* DTW = (const float*)d_in[pb + 6];
  const float* DTB = (const float*)d_in[pb + 7];
  const float* AL  = (const float*)d_in[pb + 8];
  const float* DP  = (const float*)d_in[pb + 9];
  const float* OLW = (const float*)d_in[pb + 10];
  const float* OLB = (const float*)d_in[pb + 11];
  const float* OP  = (const float*)d_in[pb + 12];

  int d  = 2 * C;                 // d_inner
  int r  = (C + 15) / 16; if (r < 1) r = 1;
  int cdim = r + 32;              // r + 2*d_state (16)
  int N2 = 2 * d;                 // in_proj out cols

  // per-k packed strides (exact per-call padded sizes)
  size_t aXp_str  = (size_t)rup(cdim, 16) * rup(d, 32) / 2;
  size_t bXs_str  = (size_t)rup(L, 64)    * rup(d, 32) / 2;
  size_t bDtr_str = (size_t)rup(L, 64)    * rup(r, 32) / 2;
  size_t aDtw_str = (size_t)rup(d, 16)    * rup(r, 32) / 2;

  k_layernorm<<<L, 256, 0, st>>>(tok, LNW, LNB, b.ln, C);
  packA(st, b.ln, b.aLn, L, C, C);
  packB(st, IP, b.bIp, N2, C, N2);
  gemm(st, b.aLn, b.bIp, nullptr, b.xz, L, N2, C, N2, 0, 0);
  k_transpose<<<g1((long)L * d), 256, 0, st>>>(b.xz, b.xxT, L, d, N2, 0);
  k_dwconv3x3<<<g1((long)d * H * W), 256, 0, st>>>(b.xxT, CW, CB, b.xxC, d, H, W, 1);
  k_build_xs<<<g1((long)d * H * W), 256, 0, st>>>(b.xxC, b.xs, d, H, W);
  for (int k = 0; k < 4; ++k) {
    packB(st, b.xs + (size_t)k * d * L, b.bXs + k * bXs_str, L, d, L);
    packA(st, XP + (size_t)k * cdim * d, b.aXp + k * aXp_str, cdim, d, d);
    gemm(st, b.aXp + k * aXp_str, b.bXs + k * bXs_str, nullptr,
         b.xdbl + (size_t)k * cdim * L, cdim, L, d, L, 0, 0);
  }
  for (int k = 0; k < 4; ++k) {
    packB(st, b.xdbl + (size_t)k * cdim * L, b.bDtr + k * bDtr_str, L, r, L);
    packA(st, DTW + (size_t)k * d * r, b.aDtw + k * aDtw_str, d, r, r);
    gemm(st, b.aDtw + k * aDtw_str, b.bDtr + k * bDtr_str, nullptr,
         b.dt + (size_t)k * d * L, d, L, r, L, 0, 0);
  }
  k_softplus_rowbias<<<g1(4L * d * L), 256, 0, st>>>(b.dt, DTB, 4 * d, L);
  k_scan_vss<<<g1(4 * d), 256, 0, st>>>(b.xs, b.dt, AL, b.xdbl, cdim, r, DP, b.ys, d, L);
  k_combine_ys<<<g1((long)d * L), 256, 0, st>>>(b.ys, b.yt, d, H, W);
  k_layernorm<<<L, 256, 0, st>>>(b.yt, OLW, OLB, b.yt, d);
  k_mul_silu_strided<<<g1((long)L * d), 256, 0, st>>>(b.yt, d, b.xz + d, N2, L, d);
  packA(st, b.yt, b.aYt, L, d, d);
  packB(st, OP, b.bOp, C, d, C);
  gemm(st, b.aYt, b.bOp, nullptr, tok, L, C, d, C, /*accum=*/1, 0);  // residual
}

// Input index map (setup_inputs dict order, params flattened recursively)
enum {
  IN_FMAP = 0, IN_LD = 1,
  E_C1D0_W = 2, E_C1D0_B = 3,
  M_INPROJ = 4, M_CONVW = 5, M_CONVB = 6, M_DTBIAS = 7, M_ALOG = 8, M_D = 9,
  M_NORMW = 10, M_OUTPROJ = 11,
  E_C1D1_W = 12, E_C1D1_B = 13,
  V32_BASE = 14,           // 13 vss params
  E_DW1W = 27, E_DW1B = 28, E_PW1W = 29, E_PW1B = 30,
  E_DW2W = 31, E_DW2B = 32, E_PW2W = 33, E_PW2B = 34,
  E_C1W = 35, E_C1B = 36, E_C2W = 37, E_C2B = 38, E_FW = 39, E_FB = 40,
  V512_BASE = 41,          // 13
  A_A1W = 54, A_A1B = 55, A_A2W = 56, A_A2B = 57,
  AVSS_BASE = 58,          // 13
  MC1_W = 71, MC1_MW = 72, MC1_MB = 73,
  MC2_W = 74, MC2_MW = 75, MC2_MB = 76
};

extern "C" void kernel_launch(void* const* d_in, const int* in_sizes, int n_in,
                              void* d_out, int out_size, void* d_ws, size_t ws_size,
                              hipStream_t stream) {
  (void)in_sizes; (void)n_in; (void)out_size; (void)ws_size;
  hipStream_t st = stream;
  const float* fmap = (const float*)d_in[IN_FMAP];
  const float* ldir = (const float*)d_in[IN_LD];
  auto F = [&](int i) { return (const float*)d_in[i]; };

  // deterministic bump allocator over d_ws
  char* base = (char*)d_ws;
  size_t off = 0;
  auto alloc  = [&](size_t bytes) -> void* { void* p = base + off; off = (off + bytes + 255) & ~(size_t)255; return p; };
  auto allocF = [&](size_t n) { return (float*)alloc(n * 4); };
  auto allocU = [&](size_t n) { return (unsigned*)alloc(n * 4); };

  // VSS workspace (sized for the 512-d / 64x64 case, reused by vss32)
  const size_t LM = 4096, CM = 512, DM = 1024;
  VssBufs vb;
  vb.ln   = allocF(LM * CM);
  vb.xz   = allocF(LM * 4 * CM);
  vb.xxT  = allocF(DM * LM);
  vb.xxC  = allocF(DM * LM);
  vb.xs   = allocF(4 * DM * LM);
  vb.xdbl = allocF(4 * 64 * LM);
  vb.dt   = allocF(4 * DM * LM);
  vb.ys   = allocF(4 * DM * LM);
  vb.yt   = allocF(LM * DM);
  vb.aLn  = allocU(LM * CM / 2);            // Mp=4096, Kp=512
  vb.bIp  = allocU((size_t)2048 * 512 / 2); // Np=2048, Kp=512
  vb.bXs  = allocU(4 * LM * DM / 2);        // 4 x (Np=4096, Kp=1024)
  vb.aXp  = allocU((size_t)4 * 64 * 1024 / 2);
  vb.bDtr = allocU((size_t)4 * 4096 * 32 / 2);
  vb.aDtw = allocU((size_t)4 * 1024 * 32 / 2);
  vb.aYt  = allocU(LM * DM / 2);
  vb.bOp  = allocU((size_t)512 * 1024 / 2);

  float* bufA = allocF((size_t)1024 * 4096);
  float* bufB = allocF((size_t)1024 * 4096);
  float* bufC = allocF((size_t)512 * 4096);
  unsigned* colP = allocU((size_t)4096 * 4608 / 2);  // packed im2col B (max)
  unsigned* wAp  = allocU((size_t)512 * 4608 / 2);   // packed weight A (max)
  unsigned* actB = allocU((size_t)4096 * 128 / 2);   // packed activation B (max)
  float* tok = allocF((size_t)4096 * 512);
  float* le  = allocF((size_t)512 * 4096);

  // ---------------- latent encoder ----------------
  float* u0 = allocF(32 * 512);
  k_conv1d<<<g1(32 * 512), 256, 0, st>>>(ldir, F(E_C1D0_W), F(E_C1D0_B), u0, 18, 32, 512, 3, 1);

  // Mamba2: L=32, D=512
  unsigned* a_u0 = allocU(32 * 512 / 2);
  unsigned* b_mip = allocU((size_t)2240 * 512 / 2);
  packA(st, u0, a_u0, 32, 512, 512);
  packB(st, F(M_INPROJ), b_mip, 2192, 512, 2192);
  float* m_zx = allocF((size_t)32 * 2192);
  gemm(st, a_u0, b_mip, nullptr, m_zx, 32, 2192, 512, 2192, 0, 0);
  float* m_dt = allocF(32 * 16);
  k_softplus_gather<<<g1(32 * 16), 256, 0, st>>>(m_zx + 2176, 2192, F(M_DTBIAS), m_dt, 32, 16);
  float* m_xbc = allocF(32 * 1152);
  k_mamba_conv<<<g1(32 * 1152), 256, 0, st>>>(m_zx, F(M_CONVW), F(M_CONVB), m_xbc, 32);
  float* m_y = allocF(32 * 1024);
  k_scan_mamba<<<4, 256, 0, st>>>(m_xbc, m_dt, F(M_ALOG), F(M_D), m_y, 32);
  float* m_yn = allocF(32 * 1024);
  k_mamba_norm<<<32, 256, 0, st>>>(m_y, m_zx, 2192, F(M_NORMW), m_yn, 1024);
  unsigned* a_myn = allocU(32 * 1024 / 2);
  unsigned* b_mop = allocU((size_t)512 * 1024 / 2);
  packA(st, m_yn, a_myn, 32, 1024, 1024);
  packB(st, F(M_OUTPROJ), b_mop, 512, 1024, 512);
  float* m_out = allocF(32 * 512);
  gemm(st, a_myn, b_mop, nullptr, m_out, 32, 512, 1024, 512, 0, 0);

  // transpose [32,512] -> [512,32]; conv1d 512 -> 1024 over L=32
  float* xT = allocF(512 * 32);
  k_transpose<<<g1(32 * 512), 256, 0, st>>>(m_out, xT, 32, 512, 512, 0);
  float* xc = allocF(1024 * 32);   // == tokens [L=1024, C=32]
  k_conv1d<<<g1(1024 * 32), 256, 0, st>>>(xT, F(E_C1D1_W), F(E_C1D1_B), xc, 512, 1024, 32, 3, 1);

  // VSS block @ d_model=32, 32x32
  run_vss(st, d_in, V32_BASE, xc, 1024, 32, 32, 32, vb);

  // tokens -> channel-major [32,1024]; dw/pw stack
  k_transpose<<<g1(1024 * 32), 256, 0, st>>>(xc, bufA, 1024, 32, 32, 0);
  k_dwconv3x3<<<g1(32 * 1024), 256, 0, st>>>(bufA, F(E_DW1W), F(E_DW1B), bufB, 32, 32, 32, 0);
  packA(st, F(E_PW1W), wAp, 64, 32, 32);
  packB(st, bufB, actB, 1024, 32, 1024);
  gemm(st, wAp, actB, F(E_PW1B), bufA, 64, 1024, 32, 1024, 0, 0);
  k_dwconv3x3<<<g1(64 * 1024), 256, 0, st>>>(bufA, F(E_DW2W), F(E_DW2B), bufB, 64, 32, 32, 0);
  packA(st, F(E_PW2W), wAp, 128, 64, 64);
  packB(st, bufB, actB, 1024, 64, 1024);
  gemm(st, wAp, actB, F(E_PW2B), bufA, 128, 1024, 64, 1024, 0, 0);

  // bilinear 32x32 -> 64x64, then conv1(3x3,128->128), conv2(1x1,128->256), final(3x3,256->512)
  k_resize<<<g1((long)128 * 64 * 64), 256, 0, st>>>(bufA, bufB, 128, 32, 32, 64, 64);
  im2colP(st, bufB, colP, 128, 64, 64);
  packA(st, F(E_C1W), wAp, 128, 1152, 1152);
  gemm(st, wAp, colP, F(E_C1B), bufA, 128, 4096, 1152, 4096, 0, 0);
  packA(st, F(E_C2W), wAp, 256, 128, 128);
  packB(st, bufA, actB, 4096, 128, 4096);
  gemm(st, wAp, actB, F(E_C2B), bufB, 256, 4096, 128, 4096, 0, 0);
  im2colP(st, bufB, colP, 256, 64, 64);
  packA(st, F(E_FW), wAp, 512, 2304, 2304);
  gemm(st, wAp, colP, F(E_FB), bufA, 512, 4096, 2304, 4096, 0, 0);

  // VSS block @ d_model=512, 64x64
  k_transpose<<<g1((long)512 * 4096), 256, 0, st>>>(bufA, tok, 512, 4096, 4096, 0);
  run_vss(st, d_in, V512_BASE, tok, 4096, 512, 64, 64, vb);
  k_transpose<<<g1((long)4096 * 512), 256, 0, st>>>(tok, le, 4096, 512, 512, 0);

  // ---------------- approximator ----------------
  k_interleave<<<g1((long)512 * 4096), 256, 0, st>>>(fmap, le, bufA, 512, 4096);
  k_groupconv2_3x3<<<g1((long)512 * 4096), 256, 0, st>>>(bufA, F(A_A1W), F(A_A1B), bufC, 512, 64, 64);
  im2colP(st, bufC, colP, 512, 64, 64);
  packA(st, F(A_A2W), wAp, 512, 4608, 4608);
  gemm(st, wAp, colP, F(A_A2B), bufB, 512, 4096, 4608, 4096, 0, 0);
  k_transpose<<<g1((long)512 * 4096), 256, 0, st>>>(bufB, tok, 512, 4096, 4096, 0);
  run_vss(st, d_in, AVSS_BASE, tok, 4096, 512, 64, 64, vb);
  k_transpose<<<g1((long)4096 * 512), 256, 0, st>>>(tok, bufA, 4096, 512, 512, 0);

  // ---------------- modulated convs ----------------
  float* style = allocF(512);
  float* svec  = allocF(512);
  k_style<<<2, 256, 0, st>>>(ldir, style);

  k_modscale<<<2, 256, 0, st>>>(style, F(MC1_MW), F(MC1_MB), svec);
  k_modweight<<<512, 256, 0, st>>>(F(MC1_W), svec, wAp, 512, 512);
  im2colP(st, bufA, colP, 512, 64, 64);
  gemm(st, wAp, colP, nullptr, bufB, 512, 4096, 4608, 4096, 0, 0);

  k_modscale<<<2, 256, 0, st>>>(style, F(MC2_MW), F(MC2_MB), svec);
  k_modweight<<<512, 256, 0, st>>>(F(MC2_W), svec, wAp, 512, 512);
  im2colP(st, bufB, colP, 512, 64, 64);
  gemm(st, wAp, colP, nullptr, (float*)d_out, 512, 4096, 4608, 4096, 0, 0);
}
